// CSHAAttention_18107582119983
// MI455X (gfx1250) — matrix operand order repaired and checked by reading, not compiled, not run on hardware
//
#include <hip/hip_runtime.h>
#include <cmath>

// ---------------------------------------------------------------------------
// CBAM-style attention (ECA channel path + 7x7 spatial path) for
// x = (B=32, C=256, H=64, W=64) fp32.  Bandwidth-bound: ~384 MiB of HBM
// traffic => ~16.5us at 23.3 TB/s.  Strategy:
//   K1  fused stats  : read x ONCE -> channel sums (per b,c) + per-pixel
//                      max/sum over channels (deterministic, no atomics)
//   K1b fold         : channel-sum partials -> per-(b,c) mean
//   K2  WMMA conv1d  : y = mean(32x256) x Band(256x256) via 8 straight-line
//                      v_wmma_f32_16x16x32_f16, band built from 5 taps
//   K3  7x7 conv     : LDS-tiled, sigmoid -> sp(b,h,w)
//   K4  apply        : out = x * (1 + ch_w[b,c] + sp[b,hw]); NT loads/stores
//                      so x (128MiB, fits 192MB L2 from K1) isn't evicted.
// ---------------------------------------------------------------------------

typedef float  v4f  __attribute__((ext_vector_type(4)));
typedef float  v8f  __attribute__((ext_vector_type(8)));
typedef _Float16 v16h __attribute__((ext_vector_type(16)));

#define B_    32
#define C_    256
#define HW_   4096
#define INV_HW (1.0f / 4096.0f)
#define INV_C  (1.0f / 256.0f)

// Fast sigmoid: v_exp_f32 + v_rcp_f32 (no IEEE div_scale chains; ~1 ulp rcp
// is far below any fp32-reference tolerance for a gating nonlinearity).
__device__ __forceinline__ float sigmoidf_(float v) {
  return __builtin_amdgcn_rcpf(1.0f + __expf(-v));
}

// ---------------------------------------------------------------------------
// K1: fused stats.  grid = 32 b * 2 chalf * 4 chunk = 256 blocks, 256 thr.
// Each thread owns 4 consecutive pixels (one float4) of its chunk and loops
// over its 128 channels: per-pixel max/sum live in registers (sole owner),
// channel partial sums are wave-reduced with __shfl_xor (wave32) and folded
// per-wave in LDS -> fully deterministic, zero atomics.
// ---------------------------------------------------------------------------
__global__ void k_stats(const float* __restrict__ x,
                        float* __restrict__ csumPart,   // [b][half][chunk][128]
                        float* __restrict__ smaxp,      // [half][b][4096]
                        float* __restrict__ ssump)      // [half][b][4096]
{
  const int tid   = threadIdx.x;
  const int blk   = blockIdx.x;
  const int chunk = blk & 3;
  const int half  = (blk >> 2) & 1;
  const int b     = blk >> 3;
  const int pos0  = chunk * 1024 + tid * 4;
  const int wave  = tid >> 5;
  const int lane  = tid & 31;

  __shared__ float csumW[8][128];
  for (int i = tid; i < 8 * 128; i += 256) (&csumW[0][0])[i] = 0.0f;
  __syncthreads();

  const float* xb = x + (size_t)(b * C_ + half * 128) * HW_ + pos0;

  v4f vmax = { -3.4e38f, -3.4e38f, -3.4e38f, -3.4e38f };
  v4f vsum = { 0.f, 0.f, 0.f, 0.f };

  for (int c = 0; c < 128; ++c) {
    const v4f v = *(const v4f*)(xb + (size_t)c * HW_);
    // Speculative prefetch (ISA: translation failure silently dropped), so
    // the final-iteration overshoot is safe; unconditional = no branch.
    __builtin_prefetch(xb + (size_t)(c + 1) * HW_, 0, 1);

    vmax.x = fmaxf(vmax.x, v.x);
    vmax.y = fmaxf(vmax.y, v.y);
    vmax.z = fmaxf(vmax.z, v.z);
    vmax.w = fmaxf(vmax.w, v.w);
    vsum += v;

    float part = (v.x + v.y) + (v.z + v.w);
    part += __shfl_xor(part, 16, 32);
    part += __shfl_xor(part, 8, 32);
    part += __shfl_xor(part, 4, 32);
    part += __shfl_xor(part, 2, 32);
    part += __shfl_xor(part, 1, 32);
    if (lane == 0) csumW[wave][c] += part;
  }

  const size_t so = ((size_t)half * B_ + b) * HW_ + pos0;
  *(v4f*)(smaxp + so) = vmax;
  *(v4f*)(ssump + so) = vsum;

  __syncthreads();
  if (tid < 128) {
    float s = 0.f;
    #pragma unroll
    for (int w = 0; w < 8; ++w) s += csumW[w][tid];
    csumPart[(((size_t)b * 2 + half) * 4 + chunk) * 128 + tid] = s;
  }
}

// ---------------------------------------------------------------------------
// K1b: fold 4 chunk partials -> per-(b,c) spatial mean.
// ---------------------------------------------------------------------------
__global__ void k_chmean(const float* __restrict__ csumPart,
                         float* __restrict__ chanMean)
{
  const int gid = blockIdx.x * blockDim.x + threadIdx.x;  // 8192
  const int b = gid >> 8, c = gid & 255;
  const int half = c >> 7, cl = c & 127;
  float s = 0.f;
  #pragma unroll
  for (int ch = 0; ch < 4; ++ch)
    s += csumPart[(((size_t)b * 2 + half) * 4 + ch) * 128 + cl];
  chanMean[gid] = s * INV_HW;
}

// ---------------------------------------------------------------------------
// K2: ECA conv1d(k=5) over channels as a banded matmul on WMMA.
//   Y(32x256) = M(32x256) x W(256x256),  W[i][j] = w1d[i-j+2] for |i-j|<=2.
// grid = 32 blocks (2 M-tiles x 16 N-tiles), 32 threads (one wave) each.
// Fragment layouts used here (A/B contiguous K, not the ISA table order):
//   A (f16 16x32): lane m = lane&15, K = e + 16*(lane>>4)  (sequential).
//   B (f16 32x16): lane n = lane&15, K = e + 16*(lane>>4)  (sequential).
//   C/D (f32 16x16): vgpr r -> m = r + 8*(lane>>4), n = lane&15.
// B is synthesized from the 5 taps, A from the channel means (f16 is ample:
// means ~ N(0, 1/4096) feeding a sigmoid; accumulate in f32).
// ---------------------------------------------------------------------------
__global__ void k_eca_wmma(const float* __restrict__ chanMean,
                           const float* __restrict__ w1d,
                           float* __restrict__ chw)
{
  const int lane = threadIdx.x;
  const int mt = blockIdx.x >> 4;    // 0..1  (batch tile)
  const int nt = blockIdx.x & 15;    // 0..15 (channel tile)
  const int hg = lane >> 4;
  const int lr = lane & 15;

  const float w0 = w1d[0], w1 = w1d[1], w2 = w1d[2], w3 = w1d[3], w4 = w1d[4];

  v8f acc = {};
  #pragma unroll
  for (int kc = 0; kc < 8; ++kc) {
    v16h a, bm;
    #pragma unroll
    for (int e = 0; e < 16; ++e) {
      // ---- A fragment: channel means ----
      const int Ka = 16 * hg + e;
      const int i  = kc * 32 + Ka;                       // input channel
      a[e] = (_Float16)chanMean[(mt * 16 + lr) * 256 + i];

      // ---- B fragment: band matrix from the 5 taps ----
      const int ib = kc * 32 + 16 * hg + e;              // row (input ch)
      const int j  = nt * 16 + lr;                       // col (output ch)
      const int d  = ib - j + 2;
      float wv = 0.0f;
      wv = (d == 0) ? w0 : wv;
      wv = (d == 1) ? w1 : wv;
      wv = (d == 2) ? w2 : wv;
      wv = (d == 3) ? w3 : wv;
      wv = (d == 4) ? w4 : wv;
      bm[e] = (_Float16)wv;
    }
    acc = __builtin_amdgcn_wmma_f32_16x16x32_f16(
        /*neg_a=*/false, a, /*neg_b=*/false, bm,
        /*c_mod=*/(short)0, acc, /*reuse_a=*/false, /*reuse_b=*/false);
  }

  #pragma unroll
  for (int r = 0; r < 8; ++r) {
    const int m = mt * 16 + r + 8 * hg;   // batch
    const int n = nt * 16 + lr;           // channel
    chw[m * 256 + n] = sigmoidf_(acc[r]); // sigmoid(conv1d) = channel weight
  }
}

// ---------------------------------------------------------------------------
// K3: 7x7 conv over [max, avg] channel maps + sigmoid.
// grid = 32 b * 8 row-bands = 256 blocks, 256 threads; LDS tile 2x14x72
// (halo handled by zero-pad), halves combined at load (max / sum*1/256).
// ---------------------------------------------------------------------------
__global__ void k_spatial_conv(const float* __restrict__ smaxp,
                               const float* __restrict__ ssump,
                               const float* __restrict__ w2,
                               const float* __restrict__ bias,
                               float* __restrict__ sp)
{
  const int tid = threadIdx.x;
  const int b   = blockIdx.x >> 3;
  const int y0  = (blockIdx.x & 7) * 8;

  __shared__ float tile[2][14][72];
  __shared__ float wk[98];
  __shared__ float bs;
  if (tid < 98) wk[tid] = w2[tid];
  if (tid == 98) bs = bias[0];

  for (int t = tid; t < 2 * 14 * 70; t += 256) {
    const int ch  = t / 980;
    const int rem = t % 980;
    const int ry = rem / 70, cx = rem % 70;
    const int gy = y0 + ry - 3, gx = cx - 3;
    float val = 0.f;
    if (gy >= 0 && gy < 64 && gx >= 0 && gx < 64) {
      const size_t off = (size_t)b * HW_ + gy * 64 + gx;
      if (ch == 0)
        val = fmaxf(smaxp[off], smaxp[(size_t)B_ * HW_ + off]);
      else
        val = (ssump[off] + ssump[(size_t)B_ * HW_ + off]) * INV_C;
    }
    tile[ch][ry][cx] = val;
  }
  __syncthreads();

  #pragma unroll
  for (int rep = 0; rep < 2; ++rep) {
    const int t  = tid + rep * 256;
    const int ry = t >> 6;      // 0..7
    const int xx = t & 63;
    float acc = bs;
    #pragma unroll
    for (int ch = 0; ch < 2; ++ch)
      #pragma unroll
      for (int ky = 0; ky < 7; ++ky)
        #pragma unroll
        for (int kx = 0; kx < 7; ++kx)
          acc += wk[ch * 49 + ky * 7 + kx] * tile[ch][ry + ky][xx + kx];
    sp[(size_t)b * HW_ + (y0 + ry) * 64 + xx] = sigmoidf_(acc);
  }
}

// ---------------------------------------------------------------------------
// K4: out = x * (1 + ch_w[b,c] + sp[b,hw]).  16 floats/thread, NT hints:
// x is last-use, out is never re-read -> don't evict the L2-resident data.
// sp/chw re-reads (512KiB / 32KiB footprints) all hit L2.
// ---------------------------------------------------------------------------
__global__ void k_apply(const float* __restrict__ x,
                        const float* __restrict__ chw,
                        const float* __restrict__ sp,
                        float* __restrict__ out)
{
  const size_t gid  = (size_t)blockIdx.x * blockDim.x + threadIdx.x;
  const size_t base = gid * 16;
  const int bc  = (int)(base >> 12);   // (b*C + c)
  const int pos = (int)(base & 4095);
  const int b   = bc >> 8;
  const float s = 1.0f + chw[bc];

  const v4f* xv  = (const v4f*)(x + base);
  const v4f* spv = (const v4f*)(sp + (size_t)b * HW_ + pos);
  v4f*       ov  = (v4f*)(out + base);

  #pragma unroll
  for (int i = 0; i < 4; ++i) {
    v4f xl = __builtin_nontemporal_load(xv + i);
    v4f sl = spv[i];
    v4f r  = xl * (sl + s);
    __builtin_nontemporal_store(r, ov + i);
  }
}

// ---------------------------------------------------------------------------
extern "C" void kernel_launch(void* const* d_in, const int* in_sizes, int n_in,
                              void* d_out, int out_size, void* d_ws, size_t ws_size,
                              hipStream_t stream)
{
  const float* x    = (const float*)d_in[0];   // (32,256,64,64)
  const float* w1   = (const float*)d_in[1];   // (1,1,5)
  const float* w2   = (const float*)d_in[2];   // (1,2,7,7)
  const float* bias = (const float*)d_in[3];   // (1,)
  float* out = (float*)d_out;

  float* ws       = (float*)d_ws;
  float* csumPart = ws;                 // 32*2*4*128        = 32768
  float* chanMean = ws + 32768;         // 32*256            =  8192
  float* chw      = ws + 40960;         // 32*256            =  8192
  float* smaxp    = ws + 49152;         // 2*32*4096         = 262144
  float* ssump    = ws + 311296;        // 2*32*4096         = 262144
  float* sp       = ws + 573440;        // 32*4096           = 131072
  (void)in_sizes; (void)n_in; (void)out_size; (void)ws_size;

  k_stats       <<<256, 256, 0, stream>>>(x, csumPart, smaxp, ssump);
  k_chmean      <<<32,  256, 0, stream>>>(csumPart, chanMean);
  k_eca_wmma    <<<32,   32, 0, stream>>>(chanMean, w1, chw);
  k_spatial_conv<<<256, 256, 0, stream>>>(smaxp, ssump, w2, bias, sp);
  k_apply       <<<8192,256, 0, stream>>>(x, chw, sp, out);
}
